// Attention_1185410974045
// MI455X (gfx1250) — matrix operand order repaired and checked by reading, not compiled, not run on hardware
//
#include <hip/hip_runtime.h>

// ---------------- problem constants (from reference) ----------------
#define TT 8192
#define HH 16
#define DD 128
#define SEQ 2048
#define NSLOTS 16384
#define HD (HH * DD)          // 2048
#define BATCH (TT / SEQ)      // 4
#define SCALE 0.08838834764831845f

// attention tiling
#define BM 64                 // queries per block
#define BN 64                 // keys per iteration
#define NWAVES 4              // 128 threads, wave32
#define QTILES (SEQ / BM)     // 32

// LDS strides (in bf16 halves), padded for bank-conflict-free b128 access
#define KBS 136               // 128 + 8  (= 256B row + 16B pad, matches TDM pad cfg)
#define VBS 72                // 64 + 8
#define PBS 72                // 64 + 8

typedef __attribute__((ext_vector_type(16))) __bf16    v16bf;
typedef __attribute__((ext_vector_type(8)))  float     v8f;
typedef __attribute__((ext_vector_type(4)))  unsigned  v4u;
typedef __attribute__((ext_vector_type(8)))  int       v8i;
typedef __attribute__((ext_vector_type(4)))  int       v4i;

#if defined(__gfx1250__) && __has_builtin(__builtin_amdgcn_tensor_load_to_lds)
#define USE_TDM 1
#else
#define USE_TDM 0
#endif

__device__ __forceinline__ unsigned short f2bf(float f) {
  unsigned u = __builtin_bit_cast(unsigned, f);
  u += 0x7fffu + ((u >> 16) & 1u);     // round-to-nearest-even
  return (unsigned short)(u >> 16);
}
__device__ __forceinline__ unsigned pack2(float a, float b) {
  return (unsigned)f2bf(a) | ((unsigned)f2bf(b) << 16);
}
__device__ __forceinline__ v16bf make_frag(uint4 a, uint4 b) {
  union { uint4 q[2]; v16bf v; } u;
  u.q[0] = a; u.q[1] = b;
  return u.v;
}

// ---------------- kernel 1: scatter K/V rows into bf16 paged cache ----------------
// Cache lives in workspace (reference's cache is an intermediate; only `o` is the
// output), so we may store it in bf16: each element converted exactly once here
// instead of ~32x inside the attention loop, and attention-side HBM reads halve.
__global__ void kv_store_kernel(const float* __restrict__ k,
                                const float* __restrict__ v,
                                const int* __restrict__ slot_mapping,
                                unsigned short* __restrict__ kcb,
                                unsigned short* __restrict__ vcb) {
  const int per_tok4 = HD / 4;                       // 512 float4 per token row
  int idx = blockIdx.x * blockDim.x + threadIdx.x;   // float4 index
  int t  = idx / per_tok4;
  int c4 = idx % per_tok4;
  if (t >= TT) return;
  int slot = slot_mapping[t];
  if (slot < 0 || slot >= NSLOTS) return;            // -1 => drop
  size_t src = (size_t)t * per_tok4 + c4;
  float4 kv4 = reinterpret_cast<const float4*>(k)[src];
  float4 vv4 = reinterpret_cast<const float4*>(v)[src];
  uint2 kp; kp.x = pack2(kv4.x, kv4.y); kp.y = pack2(kv4.z, kv4.w);
  uint2 vp; vp.x = pack2(vv4.x, vv4.y); vp.y = pack2(vv4.z, vv4.w);
  size_t dst = (size_t)slot * HD + (size_t)c4 * 4;   // in halves
  *reinterpret_cast<uint2*>(kcb + dst) = kp;
  *reinterpret_cast<uint2*>(vcb + dst) = vp;
}

#if USE_TDM
// Issue a TDM 2D tile load: 64 rows x 128 bf16, row stride HD halves, into LDS
// with hardware padding of 4 DWORDs (16B) after every 64 DWORDs (256B row) so the
// LDS row stride is KBS=136 halves. Descriptor layout per CDNA5 ISA ch.8.
__device__ __forceinline__ void tdm_load_ktile(const unsigned short* gptr,
                                               unsigned lds_off) {
  unsigned long long ga = (unsigned long long)(uintptr_t)gptr;
  v4u g0;
  g0[0] = 1u;                                        // count=1, user descriptor
  g0[1] = lds_off;                                   // lds_addr (bytes)
  g0[2] = (unsigned)ga;                              // global_addr[31:0]
  g0[3] = (unsigned)((ga >> 32) & 0x1FFFFFFu) | (2u << 30); // addr[56:32] | type=2
  v8i g1;
  g1[0] = (int)((1u << 16)       // data_size = 1 (2 bytes)
              | (1u << 20)       // pad_enable
              | (5u << 22)       // pad_interval: 2^(5+1)=64 DWORDs before pad
              | (3u << 25));     // pad_amount: 3+1 = 4 DWORDs of pad
  g1[1] = (int)(((unsigned)(HD & 0xFFFF)) << 16);    // tensor_dim0 lo16 (=2048)
  g1[2] = (int)(((unsigned)(TT & 0xFFFF)) << 16);    // dim0 hi16=0 | tensor_dim1 lo16
  g1[3] = (int)(((unsigned)DD) << 16);               // dim1 hi16=0 | tile_dim0=128
  g1[4] = BN;                                        // tile_dim1=64 | tile_dim2=0
  g1[5] = HD;                                        // tensor_dim0_stride[31:0]
  g1[6] = 0;                                         // stride hi | dim1_stride lo
  g1[7] = 0;
  v4i z4 = {0, 0, 0, 0};
#if __clang_major__ >= 23
  v8i z8 = {0, 0, 0, 0, 0, 0, 0, 0};
  __builtin_amdgcn_tensor_load_to_lds(g0, g1, z4, z4, z8, 0);
#else
  __builtin_amdgcn_tensor_load_to_lds(g0, g1, z4, z4, 0);
#endif
}
#endif

// ---------------- kernel 2: flash attention (causal) from bf16 cache ----------------
__global__ __launch_bounds__(NWAVES * 32, 1)
void attn_kernel(const float* __restrict__ q,
                 const unsigned short* __restrict__ kcb,
                 const unsigned short* __restrict__ vcb,
                 float* __restrict__ out) {
  __shared__ __align__(16) unsigned short kb[BN * KBS];          // K block [tok][d]
  __shared__ __align__(16) unsigned short vb[DD * VBS];          // V block [d][tok]
  __shared__ __align__(16) unsigned short pb[NWAVES * 16 * PBS]; // P tiles [row][col]

  const int tid  = threadIdx.x;
  const int wid  = tid >> 5;
  const int lane = tid & 31;
  const int ln16 = lane & 15;
  const int hi   = lane >> 4;          // lane half (0 or 1)

  const int qt = blockIdx.x % QTILES;
  const int h  = (blockIdx.x / QTILES) % HH;
  const int b  = blockIdx.x / (QTILES * HH);

  // ---- load Q tile rows as 4 bf16 A-fragments (16x32), SCALE pre-folded ----
  const int qrow_seq = qt * BM + wid * 16 + ln16;                 // row M = ln16
  const size_t qoff  = ((size_t)(b * SEQ + qrow_seq) * HH + h) * DD;
  v16bf qfrag[4];
#pragma unroll
  for (int kf = 0; kf < 4; ++kf) {
    const int d0 = kf * 32 + hi * 8;   // A layout: halves {d0..d0+7, d0+16..d0+23}
    float4 f0 = *reinterpret_cast<const float4*>(q + qoff + d0);
    float4 f1 = *reinterpret_cast<const float4*>(q + qoff + d0 + 4);
    float4 f2 = *reinterpret_cast<const float4*>(q + qoff + d0 + 16);
    float4 f3 = *reinterpret_cast<const float4*>(q + qoff + d0 + 20);
    uint4 a, c;
    a.x = pack2(f0.x * SCALE, f0.y * SCALE); a.y = pack2(f0.z * SCALE, f0.w * SCALE);
    a.z = pack2(f1.x * SCALE, f1.y * SCALE); a.w = pack2(f1.z * SCALE, f1.w * SCALE);
    c.x = pack2(f2.x * SCALE, f2.y * SCALE); c.y = pack2(f2.z * SCALE, f2.w * SCALE);
    c.z = pack2(f3.x * SCALE, f3.y * SCALE); c.w = pack2(f3.z * SCALE, f3.w * SCALE);
    qfrag[kf] = make_frag(a, c);
  }

  v8f acc[8];
#pragma unroll
  for (int dt = 0; dt < 8; ++dt) acc[dt] = (v8f){0.f,0.f,0.f,0.f,0.f,0.f,0.f,0.f};
  float mrow[8], lrow[8];
#pragma unroll
  for (int e = 0; e < 8; ++e) { mrow[e] = -__builtin_inff(); lrow[e] = 0.f; }

  unsigned short* pw = &pb[wid * 16 * PBS];
#if USE_TDM
  const unsigned kb_lds = (unsigned)(uintptr_t)(&kb[0]);
#endif

  const int nblocks = qt + 1;                 // causal: only key tiles <= query tile
  for (int j = 0; j < nblocks; ++j) {
    const size_t blk_tok0 = (size_t)(b * SEQ + j * BN);

#if USE_TDM
    // ---- K block staged by the Tensor Data Mover (async, padded rows) ----
    if (wid == 0)
      tdm_load_ktile(kcb + blk_tok0 * HD + h * DD, kb_lds);
#endif
    // ---- V block: bf16 copy with transpose into LDS (overlaps with TDM) ----
    {
      const int n  = tid >> 1;                // token within block
      const int d0 = (tid & 1) * 64;          // half of the 128 features
      const size_t crow = (blk_tok0 + n) * HD + h * DD + d0;   // in halves
#if !USE_TDM
#pragma unroll
      for (int c = 0; c < 64; c += 8)         // fallback K copy, no conversion
        *reinterpret_cast<uint4*>(&kb[n * KBS + d0 + c]) =
            *reinterpret_cast<const uint4*>(kcb + crow + c);
#endif
#pragma unroll
      for (int c = 0; c < 64; c += 8) {
        union { uint4 u; unsigned short s[8]; } vv;
        vv.u = *reinterpret_cast<const uint4*>(vcb + crow + c);
#pragma unroll
        for (int i = 0; i < 8; ++i)
          vb[(d0 + c + i) * VBS + n] = vv.s[i];
      }
      if (j + 1 < nblocks) // prefetch next V block -> global_prefetch_b8
        __builtin_prefetch(vcb + crow + (size_t)BN * HD, 0, 2);
    }
#if USE_TDM
    if (wid == 0) __builtin_amdgcn_s_wait_tensorcnt(0);
#endif
    __syncthreads();

    // ---- S = (Q*SCALE) * K^T  (4 N-tiles x 4 K-fragments of WMMA bf16) ----
    v8f s[4];
#pragma unroll
    for (int nt = 0; nt < 4; ++nt) {
      v8f cacc = (v8f){0.f,0.f,0.f,0.f,0.f,0.f,0.f,0.f};
      const int tok = nt * 16 + ln16;         // B layout: lane column = key token
#pragma unroll
      for (int kf = 0; kf < 4; ++kf) {
        const int base = tok * KBS + kf * 32 + hi * 8;
        uint4 u0 = *reinterpret_cast<const uint4*>(&kb[base]);
        uint4 u1 = *reinterpret_cast<const uint4*>(&kb[base + 16]);
        cacc = __builtin_amdgcn_wmma_f32_16x16x32_bf16(
            false, qfrag[kf], false, make_frag(u0, u1), (short)0, cacc, false, false);
      }
      s[nt] = cacc;
    }

    // ---- online softmax ----
    const bool diag = (j == qt);
    float mnew[8], corr[8];
#pragma unroll
    for (int e = 0; e < 8; ++e) {
      const int rowoff = wid * 16 + e + 8 * hi;     // row offset within 64-query tile
      float x = -__builtin_inff();
#pragma unroll
      for (int nt = 0; nt < 4; ++nt) {
        float vs = s[nt][e];
        const int coloff = nt * 16 + ln16;          // col offset within 64-key block
        if (diag && coloff > rowoff) vs = -__builtin_inff();
        s[nt][e] = vs;
        x = fmaxf(x, vs);
      }
#pragma unroll
      for (int m = 1; m < 16; m <<= 1) x = fmaxf(x, __shfl_xor(x, m, 32));
      mnew[e] = fmaxf(mrow[e], x);
      corr[e] = __expf(mrow[e] - mnew[e]);
    }
#pragma unroll
    for (int e = 0; e < 8; ++e) {
      float sum = 0.f;
#pragma unroll
      for (int nt = 0; nt < 4; ++nt) {
        float p = __expf(s[nt][e] - mnew[e]);
        s[nt][e] = p;
        sum += p;
      }
#pragma unroll
      for (int m = 1; m < 16; m <<= 1) sum += __shfl_xor(sum, m, 32);
      lrow[e] = lrow[e] * corr[e] + sum;
      mrow[e] = mnew[e];
    }
#pragma unroll
    for (int dt = 0; dt < 8; ++dt)
#pragma unroll
      for (int e = 0; e < 8; ++e) acc[dt][e] *= corr[e];

    // ---- write P tile (C layout -> row-major bf16 in LDS) ----
#pragma unroll
    for (int nt = 0; nt < 4; ++nt)
#pragma unroll
      for (int e = 0; e < 8; ++e)
        pw[(e + 8 * hi) * PBS + nt * 16 + ln16] = f2bf(s[nt][e]);
    __syncthreads();

    // ---- O += P * V  (8 D-tiles x 2 K-fragments) ----
    v16bf pfrag[2];
#pragma unroll
    for (int kf = 0; kf < 2; ++kf) {
      const int base = ln16 * PBS + kf * 32 + hi * 8;   // A layout from LDS
      uint4 u0 = *reinterpret_cast<const uint4*>(&pw[base]);
      uint4 u1 = *reinterpret_cast<const uint4*>(&pw[base + 16]);
      pfrag[kf] = make_frag(u0, u1);
    }
#pragma unroll
    for (int dt = 0; dt < 8; ++dt) {
      v8f cacc = acc[dt];
#pragma unroll
      for (int kf = 0; kf < 2; ++kf) {
        const int vbase = (dt * 16 + ln16) * VBS + kf * 32 + hi * 8;
        uint4 u0 = *reinterpret_cast<const uint4*>(&vb[vbase]);
        uint4 u1 = *reinterpret_cast<const uint4*>(&vb[vbase + 16]);
        cacc = __builtin_amdgcn_wmma_f32_16x16x32_bf16(
            false, pfrag[kf], false, make_frag(u0, u1), (short)0, cacc, false, false);
      }
      acc[dt] = cacc;
    }
    __syncthreads();   // protect kb/vb before next iteration's overwrite
  }

  // ---- normalize and store O ----
#pragma unroll
  for (int e = 0; e < 8; ++e) {
    const float inv = 1.0f / lrow[e];
    const int trow = b * SEQ + qt * BM + wid * 16 + e + 8 * hi;
    float* orow = out + ((size_t)trow * HH + h) * DD;
#pragma unroll
    for (int dt = 0; dt < 8; ++dt)
      orow[dt * 16 + ln16] = acc[dt][e] * inv;
  }
}

// ---------------- host launch ----------------
extern "C" void kernel_launch(void* const* d_in, const int* in_sizes, int n_in,
                              void* d_out, int out_size, void* d_ws, size_t ws_size,
                              hipStream_t stream) {
  const float* q  = (const float*)d_in[0];
  const float* k  = (const float*)d_in[1];
  const float* v  = (const float*)d_in[2];
  // d_in[3], d_in[4] are the (input) caches; inputs must not be mutated, so the
  // scatter targets a bf16 cache held in workspace instead.
  const int* slot_mapping = (const int*)d_in[5];
  float* out = (float*)d_out;

  unsigned short* kcb = (unsigned short*)d_ws;          // [NSLOTS, H*D] bf16
  unsigned short* vcb = kcb + (size_t)NSLOTS * HD;      // [NSLOTS, H*D] bf16

  // 1) scatter K/V into bf16 cache
  {
    int total4 = TT * (HD / 4);
    dim3 grid((total4 + 255) / 256), block(256);
    hipLaunchKernelGGL(kv_store_kernel, grid, block, 0, stream, k, v, slot_mapping,
                       kcb, vcb);
  }
  // 2) causal prefill attention reading from cache
  {
    dim3 grid(BATCH * HH * QTILES), block(NWAVES * 32);
    hipLaunchKernelGGL(attn_kernel, grid, block, 0, stream, q, kcb, vcb, out);
  }
  (void)in_sizes; (void)n_in; (void)out_size; (void)ws_size;
}
